// MultiheadAttention_30202210025874
// MI455X (gfx1250) — matrix operand order repaired and checked by reading, not compiled, not run on hardware
//
#include <hip/hip_runtime.h>
#include <hip/hip_bf16.h>
#include <cstdint>

// ---------------------------------------------------------------------------
// MultiheadAttention forward for MI455X (gfx1250, wave32, WMMA).
// All matmuls run through v_wmma_f32_16x16x32_bf16 (f32 accumulate).
// Block tile 128x64 (4 waves, each 32x64 = 8 accumulators), BK=32,
// double-buffered LDS so next-tile global loads overlap the WMMA chain.
// All LDS fragment gathers forced to ds_load_b128 via uint4 accesses.
// ---------------------------------------------------------------------------

typedef __attribute__((ext_vector_type(16))) __bf16 v16bf;
typedef __attribute__((ext_vector_type(8)))  float  v8f;

union Frag { v16bf v; uint32_t u[8]; uint4 q[2]; };

__device__ __forceinline__ unsigned short f2bf(float f) {
  union { float f; uint32_t u; } x; x.f = f;
  uint32_t r = x.u + 0x7FFFu + ((x.u >> 16) & 1u);   // round-to-nearest-even
  return (unsigned short)(r >> 16);
}

// ---- global -> registers (with f32->bf16 convert), then registers -> LDS ----

// A tile: 128 rows x 32 cols, one full row per thread.
template <typename T>
__device__ __forceinline__ void ldg_a(unsigned short (&ra)[32], const T* __restrict__ src,
                                      int ld, int rowBase, int k0, int tid) {
  const T* s = src + (size_t)(rowBase + tid) * ld + k0;
  if constexpr (sizeof(T) == 4) {
    const float4* s4 = (const float4*)s;
#pragma unroll
    for (int i = 0; i < 8; ++i) {
      float4 v = s4[i];
      ra[4 * i + 0] = f2bf(v.x); ra[4 * i + 1] = f2bf(v.y);
      ra[4 * i + 2] = f2bf(v.z); ra[4 * i + 3] = f2bf(v.w);
    }
  } else {
    const uint4* s4 = (const uint4*)s;
#pragma unroll
    for (int i = 0; i < 4; ++i) ((uint4*)ra)[i] = s4[i];
  }
}

__device__ __forceinline__ void sts_a(unsigned short* As, const unsigned short (&ra)[32], int tid) {
  uint4* d = (uint4*)(As + tid * 32);
#pragma unroll
  for (int i = 0; i < 4; ++i) d[i] = ((const uint4*)ra)[i];
}

// B tile: 64 rows x 32 cols in LDS as [n][k]; half a row per thread.
// BTRANS: source is [N,K] row-major.  else: source is [K,N] (transpose on load).
template <typename T, bool BTRANS>
__device__ __forceinline__ void ldg_b(unsigned short (&rb)[16], const T* __restrict__ src,
                                      int ld, int colBase, int k0, int tid) {
  const int r  = tid >> 1;          // n index 0..63
  const int kb = (tid & 1) << 4;    // k half: 0 or 16
  if constexpr (BTRANS) {
    const T* s = src + (size_t)(colBase + r) * ld + k0 + kb;
    if constexpr (sizeof(T) == 4) {
      const float4* s4 = (const float4*)s;
#pragma unroll
      for (int i = 0; i < 4; ++i) {
        float4 v = s4[i];
        rb[4 * i + 0] = f2bf(v.x); rb[4 * i + 1] = f2bf(v.y);
        rb[4 * i + 2] = f2bf(v.z); rb[4 * i + 3] = f2bf(v.w);
      }
    } else {
      const uint4* s4 = (const uint4*)s;
      ((uint4*)rb)[0] = s4[0];
      ((uint4*)rb)[1] = s4[1];
    }
  } else {
    const T* s = src + (size_t)(k0 + kb) * ld + colBase + r;
#pragma unroll
    for (int i = 0; i < 16; ++i) {
      if constexpr (sizeof(T) == 4) rb[i] = f2bf((float)s[(size_t)i * ld]);
      else                          rb[i] = (unsigned short)s[(size_t)i * ld];
    }
  }
}

__device__ __forceinline__ void sts_b(unsigned short* Bs, const unsigned short (&rb)[16], int tid) {
  const int r  = tid >> 1;
  const int kb = (tid & 1) << 4;
  uint4* d = (uint4*)(Bs + r * 32 + kb);
  d[0] = ((const uint4*)rb)[0];
  d[1] = ((const uint4*)rb)[1];
}

constexpr int EPI_BIAS_BF16 = 0;  // C = acc + bias      -> bf16
constexpr int EPI_SCORES    = 1;  // C = acc*alpha, mask -> f32
constexpr int EPI_BF16      = 2;  // C = acc             -> bf16
constexpr int EPI_BIAS_F32  = 3;  // C = acc + bias      -> f32

// C[M,N] = A[M,K] * op(B) (+bias).  BTRANS: B stored [N,K]; else [K,N].
template <typename TA, typename TB, bool BTRANS, int EPI>
__global__ __launch_bounds__(128) void gemm_wmma_kernel(
    const TA* __restrict__ A, const TB* __restrict__ Bw,
    const float* __restrict__ bias, const int* __restrict__ mask,
    void* __restrict__ Cout,
    int M, int N, int Kd, int lda, int ldb, int ldc, float alpha, int nheads,
    size_t aBS, size_t aHS, size_t bBS, size_t bHS,
    size_t cBS, size_t cHS, size_t maskBS) {
  __shared__ __align__(16) unsigned short As[2][128 * 32];   // 16 KB
  __shared__ __align__(16) unsigned short Bs[2][64 * 32];    //  8 KB

  const int zb = blockIdx.z / nheads;
  const int zh = blockIdx.z % nheads;
  A  += (size_t)zb * aBS + (size_t)zh * aHS;
  Bw += (size_t)zb * bBS + (size_t)zh * bHS;
  const size_t cOff = (size_t)zb * cBS + (size_t)zh * cHS;
  const int* maskp = nullptr;
  if constexpr (EPI == EPI_SCORES) maskp = mask ? (mask + (size_t)zb * maskBS) : nullptr;

  const int rowBase = blockIdx.y << 7;   // 128-row block tile
  const int colBase = blockIdx.x << 6;   // 64-col block tile
  const int tid  = threadIdx.x;
  const int wave = tid >> 5;
  const int lane = tid & 31;
  const int lm = lane & 15;
  const int lh = lane >> 4;

  const v8f zero = {0.f, 0.f, 0.f, 0.f, 0.f, 0.f, 0.f, 0.f};
  v8f acc[2][4] = {{zero, zero, zero, zero}, {zero, zero, zero, zero}};

  alignas(16) unsigned short ra[32];
  alignas(16) unsigned short rb[16];

  // Prologue: stage tile 0.
  ldg_a<TA>(ra, A, lda, rowBase, 0, tid);
  ldg_b<TB, BTRANS>(rb, Bw, ldb, colBase, 0, tid);
  sts_a(As[0], ra, tid);
  sts_b(Bs[0], rb, tid);
  __syncthreads();

  const int nSteps = Kd >> 5;
  for (int i = 0; i < nSteps; ++i) {
    const int cur = i & 1;
    const bool hasNext = (i + 1) < nSteps;

    // Issue next tile's global loads before the WMMA chain (latency overlap).
    if (hasNext) {
      ldg_a<TA>(ra, A, lda, rowBase, (i + 1) << 5, tid);
      ldg_b<TB, BTRANS>(rb, Bw, ldb, colBase, (i + 1) << 5, tid);
    }

    // All fragment gathers as explicit b128 LDS loads.
    const uint4* AsQ = (const uint4*)As[cur];   // row stride: 4 uint4
    const uint4* BsQ = (const uint4*)Bs[cur];

    // A fragments (16-bit A 16x32 layout): pairs pk = lh*4+(0..3) and
    // 8+lh*4+(0..3)  ->  uint4 indices row*4 + lh and row*4 + 2 + lh.
    const int r0 = ((wave << 5) + lm) << 2;
    const int r1 = r0 + (16 << 2);
    Frag a0, a1;
    a0.q[0] = AsQ[r0 + lh];
    a0.q[1] = AsQ[r0 + 2 + lh];
    a1.q[0] = AsQ[r1 + lh];
    a1.q[1] = AsQ[r1 + 2 + lh];

#pragma unroll
    for (int t = 0; t < 4; ++t) {
      // B fragment (16-bit B 32x16 layout): pairs pk = lh*4+(0..3) and
      // 8+lh*4+(0..3) -> uint4 indices row*4 + lh and row*4 + 2 + lh,  n = lm.
      const int rB = (((t << 4) + lm) << 2) + lh;
      Frag b;
      b.q[0] = BsQ[rB];
      b.q[1] = BsQ[rB + 2];
      acc[0][t] = __builtin_amdgcn_wmma_f32_16x16x32_bf16(
          false, a0.v, false, b.v, (short)0, acc[0][t], false, false);
      acc[1][t] = __builtin_amdgcn_wmma_f32_16x16x32_bf16(
          false, a1.v, false, b.v, (short)0, acc[1][t], false, false);
    }

    if (hasNext) {
      __syncthreads();                 // everyone done reading As/Bs[1-cur]
      sts_a(As[1 - cur], ra, tid);
      sts_b(Bs[1 - cur], rb, tid);
      __syncthreads();                 // next tile visible to all waves
    }
  }

  // Epilogue. C/D layout: VGPR g -> m = g + 8*lh, n = lm.
#pragma unroll
  for (int f = 0; f < 2; ++f) {
#pragma unroll
    for (int t = 0; t < 4; ++t) {
#pragma unroll
      for (int g = 0; g < 8; ++g) {
        const int gr = rowBase + (wave << 5) + (f << 4) + g + (lh << 3);
        const int gc = colBase + (t << 4) + lm;
        float val = acc[f][t][g];
        const size_t idx = cOff + (size_t)gr * ldc + gc;
        if constexpr (EPI == EPI_SCORES) {
          val *= alpha;
          if (maskp && maskp[(size_t)gr * ldc + gc] == 0) val = -3.402823466e38f;
          ((float*)Cout)[idx] = val;
        } else if constexpr (EPI == EPI_BIAS_BF16) {
          ((unsigned short*)Cout)[idx] = f2bf(val + bias[gc]);
        } else if constexpr (EPI == EPI_BF16) {
          ((unsigned short*)Cout)[idx] = f2bf(val);
        } else {
          ((float*)Cout)[idx] = val + bias[gc];
        }
      }
    }
  }
}

// Row softmax, in place. klen == 2048, 256 threads -> 8 elements/thread.
__global__ __launch_bounds__(256) void softmax_rows_kernel(float* __restrict__ p, int klen) {
  float* r = p + (size_t)blockIdx.x * klen;
  __shared__ float red[256];
  const int tid = threadIdx.x;
  float vals[8];
  float lmax = -3.402823466e38f;
#pragma unroll
  for (int i = 0; i < 8; ++i) {
    float v = r[tid + (i << 8)];
    vals[i] = v;
    lmax = fmaxf(lmax, v);
  }
  red[tid] = lmax; __syncthreads();
  for (int s = 128; s > 0; s >>= 1) {
    if (tid < s) red[tid] = fmaxf(red[tid], red[tid + s]);
    __syncthreads();
  }
  const float rmax = red[0];
  __syncthreads();
  float lsum = 0.f;
#pragma unroll
  for (int i = 0; i < 8; ++i) {
    float e = __expf(vals[i] - rmax);
    vals[i] = e;
    lsum += e;
  }
  red[tid] = lsum; __syncthreads();
  for (int s = 128; s > 0; s >>= 1) {
    if (tid < s) red[tid] += red[tid + s];
    __syncthreads();
  }
  const float inv = 1.0f / red[0];
#pragma unroll
  for (int i = 0; i < 8; ++i) r[tid + (i << 8)] = vals[i] * inv;
}

extern "C" void kernel_launch(void* const* d_in, const int* in_sizes, int n_in,
                              void* d_out, int out_size, void* d_ws, size_t ws_size,
                              hipStream_t stream) {
  (void)in_sizes; (void)n_in; (void)out_size; (void)ws_size;
  const float* key   = (const float*)d_in[0];
  const float* value = (const float*)d_in[1];
  const float* query = (const float*)d_in[2];
  const int*   mask  = (const int*)d_in[3];
  const float* wk = (const float*)d_in[4];
  const float* bk = (const float*)d_in[5];
  const float* wv = (const float*)d_in[6];
  const float* bv = (const float*)d_in[7];
  const float* wq = (const float*)d_in[8];
  const float* bq = (const float*)d_in[9];
  const float* wo = (const float*)d_in[10];
  const float* bo = (const float*)d_in[11];

  constexpr int B = 4, Q = 2048, KL = 2048, D = 1024, H = 8, DH = 128;
  constexpr float INV_SCALE = 0.08838834764831845f;  // 1/sqrt(128)

  float* out  = (float*)d_out;
  float* attw = out + (size_t)B * Q * D;             // [B,H,Q,KL] region of d_out

  unsigned short* Qbf = (unsigned short*)d_ws;       // [B*Q, D] bf16
  unsigned short* Kbf = Qbf + (size_t)B * Q  * D;    // [B*KL, D]
  unsigned short* Vbf = Kbf + (size_t)B * KL * D;    // [B*KL, D]
  unsigned short* Cbf = Qbf;                         // cvec reuses Q (dead after scores)

  const dim3 blk(128);

  // --- 1) input projections: X @ W^T + b  -> bf16 workspace -----------------
  const dim3 gProj(D / 64, (B * Q) / 128, 1);
  gemm_wmma_kernel<float, float, true, EPI_BIAS_BF16><<<gProj, blk, 0, stream>>>(
      query, wq, bq, nullptr, (void*)Qbf, B * Q, D, D, D, D, D, 1.0f, 1,
      0, 0, 0, 0, 0, 0, 0);
  gemm_wmma_kernel<float, float, true, EPI_BIAS_BF16><<<gProj, blk, 0, stream>>>(
      key, wk, bk, nullptr, (void*)Kbf, B * KL, D, D, D, D, D, 1.0f, 1,
      0, 0, 0, 0, 0, 0, 0);
  gemm_wmma_kernel<float, float, true, EPI_BIAS_BF16><<<gProj, blk, 0, stream>>>(
      value, wv, bv, nullptr, (void*)Vbf, B * KL, D, D, D, D, D, 1.0f, 1,
      0, 0, 0, 0, 0, 0, 0);

  // --- 2) scores: S = (Qh Kh^T)/sqrt(dh), masked -> attw region of d_out ----
  const dim3 gScore(KL / 64, Q / 128, B * H);
  gemm_wmma_kernel<unsigned short, unsigned short, true, EPI_SCORES><<<gScore, blk, 0, stream>>>(
      Qbf, Kbf, nullptr, mask, (void*)attw, Q, KL, DH, D, D, KL, INV_SCALE, H,
      (size_t)Q * D, (size_t)DH, (size_t)KL * D, (size_t)DH,
      (size_t)H * Q * KL, (size_t)Q * KL, (size_t)Q * KL);

  // --- 3) softmax over klen, in place ---------------------------------------
  softmax_rows_kernel<<<B * H * Q, 256, 0, stream>>>(attw, KL);

  // --- 4) context: cvec_h = attw_h @ V_h  -> bf16 workspace -----------------
  const dim3 gAV(DH / 64, Q / 128, B * H);
  gemm_wmma_kernel<float, unsigned short, false, EPI_BF16><<<gAV, blk, 0, stream>>>(
      attw, Vbf, nullptr, nullptr, (void*)Cbf, Q, DH, KL, KL, D, D, 1.0f, H,
      (size_t)H * Q * KL, (size_t)Q * KL, (size_t)KL * D, (size_t)DH,
      (size_t)Q * D, (size_t)DH, 0);

  // --- 5) output projection: out = cvec @ Wo^T + bo -> f32 d_out ------------
  gemm_wmma_kernel<unsigned short, float, true, EPI_BIAS_F32><<<gProj, blk, 0, stream>>>(
      Cbf, wo, bo, nullptr, (void*)out, B * Q, D, D, D, D, D, 1.0f, 1,
      0, 0, 0, 0, 0, 0, 0);
}
